// PLEModel_42408507081005
// MI455X (gfx1250) — matrix operand order repaired and checked by reading, not compiled, not run on hardware
//
#include <hip/hip_runtime.h>
#include <stdint.h>

typedef __attribute__((ext_vector_type(16))) __bf16 v16bf;
typedef __attribute__((ext_vector_type(8)))  float  v8f;
typedef __attribute__((ext_vector_type(4)))  int    int4x;

// Async global->LDS path (CDNA5). Guarded so compilation never regresses.
#if defined(__has_builtin)
#  if __has_builtin(__builtin_amdgcn_global_load_async_to_lds_b128) && \
      __has_builtin(__builtin_amdgcn_s_wait_asynccnt)
#    define USE_ASYNC_LDS 1
#  endif
#endif
#ifndef USE_ASYNC_LDS
#  define USE_ASYNC_LDS 0
#endif

// Builtin prototype (from hipcc diagnostic): param0 = int4 in AS1 (global),
// param1 = int4 in AS3 (LDS), then imm offset + imm cpol.
#define GPTR(p) ((__attribute__((address_space(1))) int4x*)(p))
#define LPTR(p) ((__attribute__((address_space(3))) int4x*)(p))

__device__ __forceinline__ unsigned short f2bf_u(float f) {
  union { float f; uint32_t u; } v; v.f = f;
  uint32_t r = v.u + 0x7FFFu + ((v.u >> 16) & 1u);   // round-to-nearest-even
  return (unsigned short)(r >> 16);
}
__device__ __forceinline__ float bf2f(unsigned short h) {
  union { uint32_t u; float f; } v; v.u = ((uint32_t)h) << 16;
  return v.f;
}

// ---------------------------------------------------------------------------
// 0) Weight prep: W[K,N] f32 -> Wt[N,K] bf16 (LDS-tiled transpose)
// ---------------------------------------------------------------------------
__global__ __launch_bounds__(256)
void transpose_f32_to_bf16(const float* __restrict__ W, unsigned short* __restrict__ Wt,
                           int K, int N) {
  __shared__ float tile[32][33];
  const size_t mat = (size_t)K * N;
  const float* Wz = W + (size_t)blockIdx.z * mat;
  unsigned short* Wtz = Wt + (size_t)blockIdx.z * mat;
  const int k0 = blockIdx.y * 32, n0 = blockIdx.x * 32;
  const int tx = threadIdx.x & 31, ty = threadIdx.x >> 5;    // 8 row-groups
  for (int i = ty; i < 32; i += 8)
    tile[i][tx] = Wz[(size_t)(k0 + i) * N + n0 + tx];
  __syncthreads();
  for (int i = ty; i < 32; i += 8)
    Wtz[(size_t)(n0 + i) * K + k0 + tx] = f2bf_u(tile[tx][i]);
}

// ---------------------------------------------------------------------------
// 1) Embedding gather + numerical dense -> emb [B,2176] bf16
// ---------------------------------------------------------------------------
__global__ void embed_kernel(const int* __restrict__ cat, const float* __restrict__ nx,
                             const float* __restrict__ table, const float* __restrict__ nw,
                             const float* __restrict__ nb, unsigned short* __restrict__ out,
                             int B) {
  long idx = (long)blockIdx.x * blockDim.x + threadIdx.x;
  long total = (long)B * 2176;
  if (idx >= total) return;
  int b = (int)(idx / 2176);
  int j = (int)(idx % 2176);
  float val;
  if (j < 2048) {
    int f = j >> 7, d = j & 127;
    long row = (long)cat[b * 16 + f] + (long)f * 1000;   // OFFSETS[f] = 1000*f
    val = table[row * 128 + d];
  } else {
    int d = j - 2048;
    val = nb[d];
    for (int i = 0; i < 13; ++i) val += nx[b * 13 + i] * nw[i * 128 + d];
  }
  out[idx] = f2bf_u(val);
}

// ---------------------------------------------------------------------------
// 2) Batched WMMA GEMM: C[z] = act(A[z][M,K] * Wt[z][N,K]^T + bias[z]), bf16 in,
//    f32 accum, bf16 out. Block tile 128x128, 8 waves of 64x32, K step 32,
//    double-buffered LDS fed by async global->LDS copies.
// ---------------------------------------------------------------------------
union FragU { v16bf v; uint4 q[2]; };

__global__ __launch_bounds__(256)
void wmma_gemm_bias_act(const unsigned short* __restrict__ A, long aStride,
                        const unsigned short* __restrict__ Wt, long wStride,
                        const float* __restrict__ bias, long bStride,
                        unsigned short* __restrict__ C, long cStride,
                        int M, int N, int K, int doRelu) {
  __shared__ __align__(16) unsigned short As[2][128 * 40];
  __shared__ __align__(16) unsigned short Bs[2][128 * 40];

  const int z = blockIdx.z;
  A    += (size_t)z * aStride;
  Wt   += (size_t)z * wStride;
  bias += (size_t)z * bStride;
  C    += (size_t)z * cStride;

  const int tid   = threadIdx.x;
  const int lane  = tid & 31;
  const int wv    = tid >> 5;
  const int waveM = wv >> 2;      // 0..1 -> 64 rows each
  const int waveN = wv & 3;       // 0..3 -> 32 cols each
  const int r     = lane & 15;
  const int h     = lane >> 4;

  const int m0 = blockIdx.y * 128;
  const int n0 = blockIdx.x * 128;

  // Per-thread tile-copy slots: row crow (+64), 8-half chunk cq.
  const int crow = tid >> 2, cq = tid & 3;
  const unsigned short* aSrc = A  + (size_t)(m0 + crow) * K + cq * 8;
  const unsigned short* bSrc = Wt + (size_t)(n0 + crow) * K + cq * 8;
  const size_t rowJumpA = (size_t)64 * K;
  const size_t rowJumpB = (size_t)64 * K;
  const int slot = crow * 40 + cq * 8;        // half-index in LDS tile

  v8f acc[4][2];
  for (int i = 0; i < 4; ++i)
    for (int j = 0; j < 2; ++j)
      for (int e = 0; e < 8; ++e) acc[i][j][e] = 0.0f;

  auto issue_copy = [&](int b, int koff) {
    const unsigned short* a0 = aSrc + koff;
    const unsigned short* b0 = bSrc + koff;
#if USE_ASYNC_LDS
    __builtin_amdgcn_global_load_async_to_lds_b128(GPTR(a0),            LPTR(&As[b][slot]),           0, 0);
    __builtin_amdgcn_global_load_async_to_lds_b128(GPTR(a0 + rowJumpA), LPTR(&As[b][slot + 64 * 40]), 0, 0);
    __builtin_amdgcn_global_load_async_to_lds_b128(GPTR(b0),            LPTR(&Bs[b][slot]),           0, 0);
    __builtin_amdgcn_global_load_async_to_lds_b128(GPTR(b0 + rowJumpB), LPTR(&Bs[b][slot + 64 * 40]), 0, 0);
#else
    *(uint4*)&As[b][slot]           = *(const uint4*)a0;
    *(uint4*)&As[b][slot + 64 * 40] = *(const uint4*)(a0 + rowJumpA);
    *(uint4*)&Bs[b][slot]           = *(const uint4*)b0;
    *(uint4*)&Bs[b][slot + 64 * 40] = *(const uint4*)(b0 + rowJumpB);
#endif
  };

  // Prologue: stage tile 0 into buffer 0.
  issue_copy(0, 0);
#if USE_ASYNC_LDS
  __builtin_amdgcn_s_wait_asynccnt(0);
#endif
  __syncthreads();

  int buf = 0;
  for (int k0 = 0; k0 < K; k0 += 32) {
    if (k0 + 32 < K) issue_copy(buf ^ 1, k0 + 32);   // prefetch next tile

    FragU fa[4], fb[2];
    for (int i = 0; i < 4; ++i) {
      int am = waveM * 64 + i * 16;
      // A 16x32 bf16 layout: lane-half h holds K = h*8..h*8+7 and 16+h*8..+7
      fa[i].q[0] = *(const uint4*)&As[buf][(am + r) * 40 + h * 8];
      fa[i].q[1] = *(const uint4*)&As[buf][(am + r) * 40 + 16 + h * 8];
    }
    for (int j = 0; j < 2; ++j) {
      int bn = waveN * 32 + j * 16;
      // B 32x16 bf16 layout: lane-half h holds K = h*8..h*8+7 and 16+h*8..+7
      fb[j].q[0] = *(const uint4*)&Bs[buf][(bn + r) * 40 + h * 8];
      fb[j].q[1] = *(const uint4*)&Bs[buf][(bn + r) * 40 + 16 + h * 8];
    }
    for (int i = 0; i < 4; ++i)
      for (int j = 0; j < 2; ++j)
        acc[i][j] = __builtin_amdgcn_wmma_f32_16x16x32_bf16(
            false, fa[i].v, false, fb[j].v, (short)0, acc[i][j], false, false);

#if USE_ASYNC_LDS
    __builtin_amdgcn_s_wait_asynccnt(0);   // next-tile copies landed in LDS
#endif
    __syncthreads();
    buf ^= 1;
  }

  // Epilogue: bias + (relu) + cvt bf16 store
  for (int j = 0; j < 2; ++j) {
    const int col = n0 + waveN * 32 + j * 16 + r;
    const float bj = bias[col];
    for (int i = 0; i < 4; ++i) {
      const int rowBase = m0 + waveM * 64 + i * 16 + h * 8;
      for (int e = 0; e < 8; ++e) {
        float v = acc[i][j][e] + bj;
        if (doRelu) v = v > 0.f ? v : 0.f;
        C[(size_t)(rowBase + e) * N + col] = f2bf_u(v);
      }
    }
  }
}

// ---------------------------------------------------------------------------
// 3) Gate: softmax(X[M,K](bf16) @ GW[K,NG] + GB) ; one wave per row
// ---------------------------------------------------------------------------
__global__ __launch_bounds__(256)
void gate_softmax(const unsigned short* __restrict__ X, long xStride,
                  const float* __restrict__ GW, long gwStride,
                  const float* __restrict__ GB, long gbStride,
                  float* __restrict__ G, long gStride,
                  int M, int K, int NG) {
  X  += (size_t)blockIdx.z * xStride;
  GW += (size_t)blockIdx.z * gwStride;
  GB += (size_t)blockIdx.z * gbStride;
  G  += (size_t)blockIdx.z * gStride;
  int gtid = blockIdx.x * blockDim.x + threadIdx.x;
  int row  = gtid >> 5;
  int lane = threadIdx.x & 31;
  if (row >= M) return;
  float acc[12];
  for (int g = 0; g < 12; ++g) acc[g] = 0.f;
  for (int k = lane; k < K; k += 32) {
    float x = bf2f(X[(size_t)row * K + k]);
    const float* wrow = GW + (size_t)k * NG;
    for (int g = 0; g < NG; ++g) acc[g] += x * wrow[g];
  }
  for (int off = 16; off > 0; off >>= 1)
    for (int g = 0; g < NG; ++g) acc[g] += __shfl_down(acc[g], off, 32);
  if (lane == 0) {
    float mx = -1e30f;
    for (int g = 0; g < NG; ++g) { acc[g] += GB[g]; mx = fmaxf(mx, acc[g]); }
    float s = 0.f;
    for (int g = 0; g < NG; ++g) { acc[g] = __expf(acc[g] - mx); s += acc[g]; }
    float inv = 1.f / s;
    for (int g = 0; g < NG; ++g) G[(size_t)row * NG + g] = acc[g] * inv;
  }
}

// ---------------------------------------------------------------------------
// 4) Mix: task -> 4 own experts + 4 shared; shared -> all 12
//    Expert planes ordered [t0e0..3, t1e0..3, s0..3]
// ---------------------------------------------------------------------------
__global__ void mix_task(const unsigned short* __restrict__ E, const float* __restrict__ G,
                         unsigned short* __restrict__ O, int M, int N) {
  long idx = (long)blockIdx.x * blockDim.x + threadIdx.x;
  long total = 2L * M * N;
  if (idx >= total) return;
  int n = (int)(idx % N);
  long bn = idx / N;
  int b = (int)(bn % M);
  int t = (int)(bn / M);
  const float* g = G + ((size_t)t * M + b) * 8;
  const size_t plane = (size_t)M * N;
  const size_t off = (size_t)b * N + n;
  float v = 0.f;
  for (int k = 0; k < 4; ++k) v += g[k]     * bf2f(E[(size_t)(t * 4 + k) * plane + off]);
  for (int k = 0; k < 4; ++k) v += g[4 + k] * bf2f(E[(size_t)(8 + k)     * plane + off]);
  O[idx] = f2bf_u(v);
}

__global__ void mix_shared(const unsigned short* __restrict__ E, const float* __restrict__ GS,
                           unsigned short* __restrict__ O, int M, int N) {
  long idx = (long)blockIdx.x * blockDim.x + threadIdx.x;
  long total = (long)M * N;
  if (idx >= total) return;
  int b = (int)(idx / N);
  const float* g = GS + (size_t)b * 12;
  const size_t plane = (size_t)M * N;
  float v = 0.f;
  for (int k = 0; k < 12; ++k) v += g[k] * bf2f(E[(size_t)k * plane + idx]);
  O[idx] = f2bf_u(v);
}

// ---------------------------------------------------------------------------
// 5) Tower output: sigmoid(h[128] . wo + bo) ; one wave per (t,b)
// ---------------------------------------------------------------------------
__global__ __launch_bounds__(256)
void tower_out(const unsigned short* __restrict__ H, const float* __restrict__ WO,
               const float* __restrict__ BO, float* __restrict__ out, int M) {
  int gtid = blockIdx.x * blockDim.x + threadIdx.x;
  int w = gtid >> 5;                 // 0 .. 2*M-1  (= t*M + b)
  int lane = threadIdx.x & 31;
  if (w >= 2 * M) return;
  int t = w / M;
  const unsigned short* h = H + (size_t)w * 128;
  const float* wo = WO + t * 128;
  float s = 0.f;
  for (int i = lane; i < 128; i += 32) s += bf2f(h[i]) * wo[i];
  for (int off = 16; off > 0; off >>= 1) s += __shfl_down(s, off, 32);
  if (lane == 0) out[w] = 1.f / (1.f + __expf(-(s + BO[t])));
}

// ---------------------------------------------------------------------------
// Host orchestration
// ---------------------------------------------------------------------------
extern "C" void kernel_launch(void* const* d_in, const int* in_sizes, int n_in,
                              void* d_out, int out_size, void* d_ws, size_t ws_size,
                              hipStream_t stream) {
  const int B = 8192;
  const int*   cat    = (const int*)d_in[0];
  const float* nx     = (const float*)d_in[1];
  const float* table  = (const float*)d_in[2];
  const float* num_w  = (const float*)d_in[3];
  const float* num_b  = (const float*)d_in[4];
  const float* sw0    = (const float*)d_in[5];
  const float* sb0    = (const float*)d_in[6];
  const float* tw0    = (const float*)d_in[7];
  const float* tb0    = (const float*)d_in[8];
  const float* tgw0   = (const float*)d_in[9];
  const float* tgb0   = (const float*)d_in[10];
  const float* sgw0   = (const float*)d_in[11];
  const float* sgb0   = (const float*)d_in[12];
  const float* sw1    = (const float*)d_in[13];
  const float* sb1    = (const float*)d_in[14];
  const float* tw1    = (const float*)d_in[15];
  const float* tb1    = (const float*)d_in[16];
  const float* tgw1   = (const float*)d_in[17];
  const float* tgb1   = (const float*)d_in[18];
  // d_in[19], d_in[20] (sgw1, sgb1) unused: last PLE layer keeps shared as-is
  const float* tow_w0 = (const float*)d_in[21];
  const float* tow_b0 = (const float*)d_in[22];
  const float* tow_w1 = (const float*)d_in[23];
  const float* tow_b1 = (const float*)d_in[24];
  const float* tow_wo = (const float*)d_in[25];
  const float* tow_bo = (const float*)d_in[26];

  char* ws = (char*)d_ws;
  size_t off = 0;
  auto alloc = [&](size_t bytes) -> void* {
    void* p = ws + off;
    off = (off + bytes + 255) & ~(size_t)255;
    return p;
  };
  unsigned short* embBF  = (unsigned short*)alloc((size_t)B * 2176 * 2);
  unsigned short* eout0  = (unsigned short*)alloc((size_t)12 * B * 1024 * 2);
  float*          g0     = (float*)alloc((size_t)2 * B * 8 * 4);
  float*          gs0    = (float*)alloc((size_t)B * 12 * 4);
  unsigned short* x1t    = (unsigned short*)alloc((size_t)2 * B * 1024 * 2);
  unsigned short* x1s    = (unsigned short*)alloc((size_t)B * 1024 * 2);
  unsigned short* eout1  = (unsigned short*)alloc((size_t)12 * B * 512 * 2);
  float*          g1     = (float*)alloc((size_t)2 * B * 8 * 4);
  unsigned short* x2t    = (unsigned short*)alloc((size_t)2 * B * 512 * 2);
  unsigned short* h0     = (unsigned short*)alloc((size_t)2 * B * 256 * 2);
  unsigned short* h1     = (unsigned short*)alloc((size_t)2 * B * 128 * 2);
  // transposed bf16 weights [N,K]
  unsigned short* tw0t   = (unsigned short*)alloc((size_t)8 * 2176 * 1024 * 2);
  unsigned short* sw0t   = (unsigned short*)alloc((size_t)4 * 2176 * 1024 * 2);
  unsigned short* tw1t   = (unsigned short*)alloc((size_t)8 * 1024 * 512 * 2);
  unsigned short* sw1t   = (unsigned short*)alloc((size_t)4 * 1024 * 512 * 2);
  unsigned short* tow0t  = (unsigned short*)alloc((size_t)2 * 512 * 256 * 2);
  unsigned short* tow1t  = (unsigned short*)alloc((size_t)2 * 256 * 128 * 2);

  dim3 blk(256);

  // Weight prep (once per launch; graph-safe)
  transpose_f32_to_bf16<<<dim3(1024 / 32, 2176 / 32, 8), blk, 0, stream>>>(tw0, tw0t, 2176, 1024);
  transpose_f32_to_bf16<<<dim3(1024 / 32, 2176 / 32, 4), blk, 0, stream>>>(sw0, sw0t, 2176, 1024);
  transpose_f32_to_bf16<<<dim3(512 / 32, 1024 / 32, 8), blk, 0, stream>>>(tw1, tw1t, 1024, 512);
  transpose_f32_to_bf16<<<dim3(512 / 32, 1024 / 32, 4), blk, 0, stream>>>(sw1, sw1t, 1024, 512);
  transpose_f32_to_bf16<<<dim3(256 / 32, 512 / 32, 2), blk, 0, stream>>>(tow_w0, tow0t, 512, 256);
  transpose_f32_to_bf16<<<dim3(128 / 32, 256 / 32, 2), blk, 0, stream>>>(tow_w1, tow1t, 256, 128);

  // Embedding
  long embTotal = (long)B * 2176;
  embed_kernel<<<dim3((unsigned)((embTotal + 255) / 256)), blk, 0, stream>>>(
      cat, nx, table, num_w, num_b, embBF, B);

  // --- PLE layer 0 (input = emb for both tasks and shared) ---
  wmma_gemm_bias_act<<<dim3(1024 / 128, B / 128, 8), blk, 0, stream>>>(
      embBF, 0, tw0t, (long)2176 * 1024, tb0, 1024,
      eout0, (long)B * 1024, B, 1024, 2176, 1);
  wmma_gemm_bias_act<<<dim3(1024 / 128, B / 128, 4), blk, 0, stream>>>(
      embBF, 0, sw0t, (long)2176 * 1024, sb0, 1024,
      eout0 + (size_t)8 * B * 1024, (long)B * 1024, B, 1024, 2176, 1);
  gate_softmax<<<dim3(B * 32 / 256, 1, 2), blk, 0, stream>>>(
      embBF, 0, tgw0, (long)2176 * 8, tgb0, 8, g0, (long)B * 8, B, 2176, 8);
  gate_softmax<<<dim3(B * 32 / 256, 1, 1), blk, 0, stream>>>(
      embBF, 0, sgw0, 0, sgb0, 0, gs0, 0, B, 2176, 12);
  mix_task<<<dim3((unsigned)((2L * B * 1024 + 255) / 256)), blk, 0, stream>>>(
      eout0, g0, x1t, B, 1024);
  mix_shared<<<dim3((unsigned)(((long)B * 1024 + 255) / 256)), blk, 0, stream>>>(
      eout0, gs0, x1s, B, 1024);

  // --- PLE layer 1 (last=True) ---
  wmma_gemm_bias_act<<<dim3(512 / 128, B / 128, 4), blk, 0, stream>>>(
      x1t, 0, tw1t, (long)1024 * 512, tb1, 512,
      eout1, (long)B * 512, B, 512, 1024, 1);
  wmma_gemm_bias_act<<<dim3(512 / 128, B / 128, 4), blk, 0, stream>>>(
      x1t + (size_t)B * 1024, 0, tw1t + (size_t)4 * 1024 * 512, (long)1024 * 512,
      tb1 + 4 * 512, 512, eout1 + (size_t)4 * B * 512, (long)B * 512, B, 512, 1024, 1);
  wmma_gemm_bias_act<<<dim3(512 / 128, B / 128, 4), blk, 0, stream>>>(
      x1s, 0, sw1t, (long)1024 * 512, sb1, 512,
      eout1 + (size_t)8 * B * 512, (long)B * 512, B, 512, 1024, 1);
  gate_softmax<<<dim3(B * 32 / 256, 1, 2), blk, 0, stream>>>(
      x1t, (long)B * 1024, tgw1, (long)1024 * 8, tgb1, 8, g1, (long)B * 8, B, 1024, 8);
  mix_task<<<dim3((unsigned)((2L * B * 512 + 255) / 256)), blk, 0, stream>>>(
      eout1, g1, x2t, B, 512);

  // --- Towers ---
  wmma_gemm_bias_act<<<dim3(256 / 128, B / 128, 2), blk, 0, stream>>>(
      x2t, (long)B * 512, tow0t, (long)512 * 256, tow_b0, 256,
      h0, (long)B * 256, B, 256, 512, 1);
  wmma_gemm_bias_act<<<dim3(128 / 128, B / 128, 2), blk, 0, stream>>>(
      h0, (long)B * 256, tow1t, (long)256 * 128, tow_b1, 128,
      h1, (long)B * 128, B, 128, 256, 1);
  tower_out<<<dim3(2 * B * 32 / 256), blk, 0, stream>>>(
      h1, tow_wo, tow_bo, (float*)d_out, B);
}
